// MACEModel_54262616818157
// MI455X (gfx1250) — matrix operand order repaired and checked by reading, not compiled, not run on hardware
//
#include <hip/hip_runtime.h>
#include <complex>
#include <cmath>
#include <cstring>
#include <algorithm>

// ---------------- model constants ----------------
#define E_    64000
#define N_    8000
#define MULC  16
#define NBC   8
#define MLPC  128
#define ENCC  256
#define SHDC  9

typedef __attribute__((ext_vector_type(16))) _Float16 v16h;
typedef __attribute__((ext_vector_type(8)))  float    v8f;

// compile-time path tables (PATHS0 is the first 3 entries of PATHS1)
constexpr int PL1[11] = {0,0,0,1,1,1,1,2,2,2,2};
constexpr int PL2[11] = {0,1,2,0,1,1,2,0,1,2,2};
constexpr int PL3[11] = {0,1,2,1,0,2,1,2,1,0,2};
constexpr int OFFA[3] = {0,1,4};
constexpr int DIMA[3] = {1,3,5};
// layer0: npth={1,1,1} -> 1/sqrt(16);  layer1: npth={3,4,4} -> 1/sqrt(48), 1/sqrt(64)
constexpr float NRM0[11] = {0.25f,0.25f,0.25f,0.f,0.f,0.f,0.f,0.f,0.f,0.f,0.f};
constexpr float NRM1[11] = {0.14433756729740643f,0.125f,0.125f,0.125f,
                            0.14433756729740643f,0.125f,0.125f,0.125f,
                            0.125f,0.14433756729740643f,0.125f};

struct CGTab { float v[9 * 9 * 9]; };   // full 9x9x9 CG tensor (disjoint path blocks)

// ---------------- kernels ----------------

__global__ void zero_kernel(float* __restrict__ p, int count) {
    int i = blockIdx.x * blockDim.x + threadIdx.x;
    if (i < count) p[i] = 0.f;
}

// per-edge geometry: spherical harmonics (9) + radial bessel basis (8)
__global__ void geom_kernel(const float* __restrict__ pos, const int* __restrict__ eidx,
                            float* __restrict__ sh, float* __restrict__ ef) {
    int e = blockIdx.x * blockDim.x + threadIdx.x;
    if (e >= E_) return;
    int s = eidx[e], d = eidx[E_ + e];
    float vx = pos[s * 3 + 0] - pos[d * 3 + 0];
    float vy = pos[s * 3 + 1] - pos[d * 3 + 1];
    float vz = pos[s * 3 + 2] - pos[d * 3 + 2];
    float r = sqrtf(vx * vx + vy * vy + vz * vz);
    r = fmaxf(r, 1e-6f);
    float ux = vx / r, uy = vy / r, uz = vz / r;
    const float s3 = 1.7320508075688772f, s15 = 3.872983346207417f, s5 = 2.23606797749979f;
    float* shp = sh + (size_t)e * 9;
    shp[0] = 1.f;            shp[1] = s3 * uy;       shp[2] = s3 * uz;  shp[3] = s3 * ux;
    shp[4] = s15 * ux * uy;  shp[5] = s15 * uy * uz; shp[6] = 0.5f * s5 * (3.f * uz * uz - 1.f);
    shp[7] = s15 * ux * uz;  shp[8] = 0.5f * s15 * (ux * ux - uy * uy);
    float uu = r * 0.1f;                              // r / RMAX
    float env = 0.f;
    if (uu < 1.f) {
        float u5 = uu * uu * uu * uu * uu;
        env = 1.f - 21.f * u5 + 35.f * u5 * uu - 15.f * u5 * uu * uu;  // P=5 cutoff poly
    }
    float pref = sqrtf(0.2f) / r * env;               // sqrt(2/RMAX)/r * env
    float* efp = ef + (size_t)e * 8;
#pragma unroll
    for (int nb = 1; nb <= 8; nb++)
        efp[nb - 1] = pref * __sinf((float)nb * 3.14159265358979323846f * r * 0.1f);
}

// X0[n][u][c] : c==0 gets embedding, rest zero (doubles as sc0)
__global__ void embed_kernel(const int* __restrict__ atoms, const float* __restrict__ emb,
                             float* __restrict__ X0) {
    int idx = blockIdx.x * blockDim.x + threadIdx.x;
    if (idx >= N_ * 144) return;
    int n = idx / 144, rem = idx % 144, u = rem / 9, c = rem % 9;
    X0[idx] = (c == 0) ? emb[atoms[n] * MULC + u] : 0.f;
}

// hidden = relu(ef @ f1w + f1b), stored f16 (K=8, VALU is fine)
__global__ void mlp_kernel(const float* __restrict__ ef, const float* __restrict__ w,
                           const float* __restrict__ b, _Float16* __restrict__ hid) {
    int idx = blockIdx.x * blockDim.x + threadIdx.x;
    if (idx >= E_ * MLPC) return;
    int e = idx >> 7, m = idx & 127;
    float a = b[m];
#pragma unroll
    for (int k = 0; k < NBC; k++) a += ef[e * 8 + k] * w[k * MLPC + m];
    hid[idx] = (_Float16)fmaxf(a, 0.f);
}

// pack f2w (f32, MLP x WN row-major) into WMMA-B f16 fragment layout:
// fragment (nt, ks, lane) holds B[K = ks*32 + 8*(lane>>4) + j + (j>=8 ? 8 : 0)][col = nt*16 + (lane&15)], j=0..15
__global__ void packB_kernel(const float* __restrict__ w, _Float16* __restrict__ Bp, int WN) {
    int idx = blockIdx.x * blockDim.x + threadIdx.x;
    int total = (WN / 16) * 4 * 32;
    if (idx >= total) return;
    int lane = idx & 31, ks = (idx >> 5) & 3, nt = idx >> 7;
    int col = nt * 16 + (lane & 15);
    int kbase = ks * 32 + ((lane >> 4) << 3);
    _Float16* o = Bp + (size_t)idx * 16;
#pragma unroll
    for (int j = 0; j < 16; j++) o[j] = (_Float16)w[(size_t)(kbase + j + (j >= 8 ? 8 : 0)) * WN + col];
}

// fused conv: per 16-edge tile, WMMA-computed per-edge weight blocks applied to CG-contracted
// features; register accumulation across all paths, one LDS reduce, then global segment-sum.
template <int LAYER>
__global__ __launch_bounds__(128) void conv_kernel(
    const _Float16* __restrict__ hid, const _Float16* __restrict__ Bp,
    const float* __restrict__ f2b, const float* __restrict__ X,
    const float* __restrict__ sh, const int* __restrict__ eidx,
    float* __restrict__ Aout, CGTab cg) {
    constexpr int NP = (LAYER == 0) ? 3 : 11;
    __shared__ float x_t[16][16][9];
    __shared__ float sh_t[16][9];
    __shared__ float out_t[16][16][9];
    __shared__ float tmp_t[16][16][5];
    __shared__ int   nsrc[16], ndst[16];

    const int tid = threadIdx.x;
    const int lane = tid & 31, wave = tid >> 5;
    const int e0 = blockIdx.x * 16;

    if (tid < 16) { nsrc[tid] = eidx[e0 + tid]; ndst[tid] = eidx[E_ + e0 + tid]; }
    for (int i = tid; i < 16 * 9; i += 128) ((float*)sh_t)[i] = sh[(size_t)e0 * 9 + i];
    for (int i = tid; i < 16 * 16 * 9; i += 128) ((float*)out_t)[i] = 0.f;
    __syncthreads();
    for (int i = tid; i < 16 * 16 * 9; i += 128) {
        int e = i / 144, rem = i % 144;
        ((float*)x_t)[i] = X[(size_t)ndst[e] * 144 + rem];
    }

    // A fragments: hidden tile 16x128 f16, WMMA 16x16x32 A layout
    v16h afrag[4];
    const int M = lane & 15, kh = lane >> 4;
#pragma unroll
    for (int ks = 0; ks < 4; ks++) {
#pragma unroll
        for (int t = 0; t < 8; t++) {
            int baseK = ks * 32 + (t < 4 ? 2 * t : 16 + 2 * (t - 4)) + 8 * kh;
            afrag[ks][2 * t]     = hid[(size_t)(e0 + M) * MLPC + baseK];
            afrag[ks][2 * t + 1] = hid[(size_t)(e0 + M) * MLPC + baseK + 1];
        }
    }
    __syncthreads();

    // per-lane accumulator over ALL paths: accv[r][c], c = SHD channel
    float accv[8][9];
#pragma unroll
    for (int r = 0; r < 8; r++)
#pragma unroll
        for (int c = 0; c < 9; c++) accv[r][c] = 0.f;

#pragma unroll
    for (int p = 0; p < NP; p++) {
        constexpr const int* L1A = PL1;  // keep constexpr context
        const int L1 = L1A[p], L2 = PL2[p], L3 = PL3[p];
        const int o1 = OFFA[L1], o2 = OFFA[L2], o3 = OFFA[L3];
        const int d1 = DIMA[L1], d2 = DIMA[L2], d3 = DIMA[L3];
        const float nrm = (LAYER == 0) ? NRM0[p] : NRM1[p];

        // tmp[e][u][k] = nrm * sum_ij x[dst[e]][u][o1+i] * sh[e][o2+j] * CG[i][j][k]
#pragma unroll
        for (int pair = tid; pair < 256; pair += 128) {
            int e = pair >> 4, u = pair & 15;
            float acc[5];
#pragma unroll
            for (int k = 0; k < 5; k++) acc[k] = 0.f;
#pragma unroll
            for (int i = 0; i < 5; i++) {
                if (i < d1) {
                    float xv = x_t[e][u][o1 + i];
#pragma unroll
                    for (int j = 0; j < 5; j++) {
                        if (j < d2) {
                            float sv = xv * sh_t[e][o2 + j];
                            const float* cgp = &cg.v[((o1 + i) * 9 + (o2 + j)) * 9 + o3];
#pragma unroll
                            for (int k = 0; k < 5; k++)
                                if (k < d3) acc[k] += sv * cgp[k];
                        }
                    }
                }
            }
#pragma unroll
            for (int k = 0; k < 5; k++)
                if (k < d3) tmp_t[e][u][k] = acc[k] * nrm;
        }
        __syncthreads();

        // ww tiles via WMMA; wave owns u-tiles t = wave*4 .. wave*4+3
#pragma unroll
        for (int tl = 0; tl < 4; tl++) {
            const int t = wave * 4 + tl;               // u index of this tile
            const int nt = p * 16 + t;                 // global column tile in f2w
            float bias = f2b[p * 256 + t * 16 + (lane & 15)];
            v8f c;
#pragma unroll
            for (int r = 0; r < 8; r++) c[r] = bias;
#pragma unroll
            for (int ks = 0; ks < 4; ks++) {
                const _Float16* bp = Bp + ((size_t)(nt * 4 + ks) * 32 + lane) * 16;
                __builtin_prefetch(bp + 512, 0, 1);
                v16h bf;
#pragma unroll
                for (int j = 0; j < 16; j++) bf[j] = bp[j];
                c = __builtin_amdgcn_wmma_f32_16x16x32_f16(false, afrag[ks], false, bf,
                                                           (short)0, c, false, false);
            }
            // lane holds ww[e = r+8*kh][u=t][v = lane&15]
#pragma unroll
            for (int r = 0; r < 8; r++) {
                int e = r + 8 * kh;
                float wwv = c[r];
#pragma unroll
                for (int k = 0; k < 5; k++)
                    if (k < d3) accv[r][o3 + k] += tmp_t[e][t][k] * wwv;
            }
        }
        __syncthreads();   // tmp_t safe to overwrite next path
    }

    // one LDS reduce across the 4 waves (each wave covered a disjoint u-range)
    {
        const int v = lane & 15;
#pragma unroll
        for (int r = 0; r < 8; r++) {
            int e = r + 8 * kh;
#pragma unroll
            for (int c = 0; c < 9; c++) atomicAdd(&out_t[e][v][c], accv[r][c]);
        }
    }
    __syncthreads();

    // segment_sum into nodes
    for (int i = tid; i < 16 * 16 * 9; i += 128) {
        int e = i / 144, rem = i % 144;
        atomicAdd(&Aout[(size_t)nsrc[e] * 144 + rem], ((float*)out_t)[i]);
    }
}

// per-node symmetric tensor product (CORR=3) + channel mixing + skip connection
__global__ __launch_bounds__(128) void prod_kernel(
    const float* __restrict__ Ain, const float* __restrict__ sc,
    const float* __restrict__ pw, const float* __restrict__ lw,
    float* __restrict__ Hout, CGTab cgf) {
    __shared__ float sout[8][16][9];
    int tid = threadIdx.x;
    int nl = tid >> 4, u = tid & 15;
    size_t n = (size_t)blockIdx.x * 8 + nl;
    float xn[9], m[9], outv[9];
    const float* ap = Ain + n * 144 + (size_t)u * 9;
#pragma unroll
    for (int c = 0; c < 9; c++) { xn[c] = ap[c]; m[c] = xn[c]; }
#pragma unroll
    for (int l = 0; l < 3; l++) {
        float w = pw[(0 * 3 + l) * MULC + u];
#pragma unroll
        for (int k = 0; k < 5; k++)
            if (k < DIMA[l]) outv[OFFA[l] + k] = w * m[OFFA[l] + k];
    }
#pragma unroll
    for (int nu = 1; nu < 3; nu++) {
        float mn[9];
#pragma unroll
        for (int k = 0; k < 9; k++) mn[k] = 0.f;
#pragma unroll
        for (int i = 0; i < 9; i++) {
#pragma unroll
            for (int j = 0; j < 9; j++) {
                float pp = m[i] * xn[j];
                const float* row = &cgf.v[(i * 9 + j) * 9];
#pragma unroll
                for (int k = 0; k < 9; k++) mn[k] += pp * row[k];
            }
        }
#pragma unroll
        for (int k = 0; k < 9; k++) m[k] = mn[k];
#pragma unroll
        for (int l = 0; l < 3; l++) {
            float w = pw[(nu * 3 + l) * MULC + u];
#pragma unroll
            for (int k = 0; k < 5; k++)
                if (k < DIMA[l]) outv[OFFA[l] + k] += w * m[OFFA[l] + k];
        }
    }
#pragma unroll
    for (int c = 0; c < 9; c++) sout[nl][u][c] = outv[c];
    __syncthreads();
    int v = u;
#pragma unroll
    for (int l = 0; l < 3; l++) {
#pragma unroll
        for (int k = 0; k < 5; k++) {
            if (k < DIMA[l]) {
                float acc = 0.f;
#pragma unroll
                for (int uu = 0; uu < 16; uu++)
                    acc += sout[nl][uu][OFFA[l] + k] * lw[(l * MULC + uu) * MULC + v];
                size_t idx = n * 144 + (size_t)v * 9 + OFFA[l] + k;
                Hout[idx] = acc * 0.25f + sc[idx];
            }
        }
    }
}

// enc = h[:,:,0] @ enc_w / sqrt(MUL), WMMA with K padded 16->32.
// Branch-free zero padding: clamp the index so loads are always in-bounds,
// then select 0 with v_cndmask (no EXEC manipulation around loads).
__global__ __launch_bounds__(32) void enc_kernel(const float* __restrict__ H,
                                                 const float* __restrict__ ew,
                                                 float* __restrict__ out) {
    int lane = threadIdx.x;
    int n0 = blockIdx.x * 16;
    int nt = blockIdx.y;
    int Mr = lane & 15, kh = lane >> 4;
    v16h a, b;
#pragma unroll
    for (int t = 0; t < 8; t++) {
        int k0 = (t < 4 ? 2 * t : 16 + 2 * (t - 4)) + 8 * kh;
        float h0 = H[(size_t)(n0 + Mr) * 144 + (size_t)(k0 & 15) * 9];
        float h1 = H[(size_t)(n0 + Mr) * 144 + (size_t)((k0 + 1) & 15) * 9];
        a[2 * t]     = (_Float16)((k0 < 16)     ? h0 : 0.f);
        a[2 * t + 1] = (_Float16)((k0 + 1 < 16) ? h1 : 0.f);
    }
    int col = nt * 16 + Mr;
#pragma unroll
    for (int j = 0; j < 16; j++) {
        int K = 8 * kh + j + (j >= 8 ? 8 : 0);
        float bv = ew[(K & 15) * ENCC + col];
        b[j] = (_Float16)((K < 16) ? bv : 0.f);
    }
    v8f c = {};
    c = __builtin_amdgcn_wmma_f32_16x16x32_f16(false, a, false, b, (short)0, c, false, false);
#pragma unroll
    for (int r = 0; r < 8; r++) {
        int row = n0 + r + 8 * kh;
        out[(size_t)row * ENCC + col] = c[r] * 0.25f;
    }
}

// ---------------- host: Clebsch-Gordan tables (mirrors reference) ----------------
namespace hcg {
static double fct(int n) { double r = 1; for (int i = 2; i <= n; i++) r *= i; return r; }

static double cg_c(int j1, int m1, int j2, int m2, int j3, int m3) {
    if (m1 + m2 != m3) return 0.0;
    if (j3 < std::abs(j1 - j2) || j3 > j1 + j2) return 0.0;
    double pre = std::sqrt((2 * j3 + 1) * fct(j3 + j1 - j2) * fct(j3 - j1 + j2) *
                           fct(j1 + j2 - j3) / fct(j1 + j2 + j3 + 1));
    pre *= std::sqrt(fct(j3 + m3) * fct(j3 - m3) * fct(j1 - m1) * fct(j1 + m1) *
                     fct(j2 - m2) * fct(j2 + m2));
    double s = 0.0;
    for (int k = 0; k <= j1 + j2 + j3; k++) {
        int d0 = k, d1 = j1 + j2 - j3 - k, d2 = j1 - m1 - k, d3 = j2 + m2 - k,
            d4 = j3 - j2 + m1 + k, d5 = j3 - j1 - m2 + k;
        if (d0 < 0 || d1 < 0 || d2 < 0 || d3 < 0 || d4 < 0 || d5 < 0) continue;
        s += ((k & 1) ? -1.0 : 1.0) /
             (fct(d0) * fct(d1) * fct(d2) * fct(d3) * fct(d4) * fct(d5));
    }
    return pre * s;
}

static void U_mat(int l, std::complex<double> U[5][5]) {
    for (int i = 0; i < 5; i++) for (int j = 0; j < 5; j++) U[i][j] = 0.0;
    const double is2 = 1.0 / std::sqrt(2.0);
    U[l][l] = 1.0;
    for (int m = 1; m <= l; m++) {
        double sgn = (m & 1) ? -1.0 : 1.0;
        U[l + m][l - m] = is2;
        U[l + m][l + m] = sgn * is2;
        U[l - m][l - m] = std::complex<double>(0.0, is2);
        U[l - m][l + m] = std::complex<double>(0.0, -sgn * is2);
    }
}

static void real_w3j(int l1, int l2, int l3, double T[5][5][5]) {
    int d1 = 2 * l1 + 1, d2 = 2 * l2 + 1, d3 = 2 * l3 + 1;
    std::complex<double> C[5][5][5];
    for (int a = 0; a < d1; a++)
        for (int b = 0; b < d2; b++)
            for (int c = 0; c < d3; c++)
                C[a][b][c] = cg_c(l1, a - l1, l2, b - l2, l3, c - l3);
    std::complex<double> U1[5][5], U2[5][5], U3[5][5];
    U_mat(l1, U1); U_mat(l2, U2); U_mat(l3, U3);
    double Tr[5][5][5], Ti[5][5][5], mr = 0.0, mi = 0.0;
    for (int i = 0; i < d1; i++)
        for (int j = 0; j < d2; j++)
            for (int k = 0; k < d3; k++) {
                std::complex<double> s = 0.0;
                for (int a = 0; a < d1; a++)
                    for (int b = 0; b < d2; b++)
                        for (int c = 0; c < d3; c++)
                            s += U1[i][a] * U2[j][b] * std::conj(U3[k][c]) * C[a][b][c];
                Tr[i][j][k] = s.real(); Ti[i][j][k] = s.imag();
                mr = std::max(mr, std::fabs(s.real()));
                mi = std::max(mi, std::fabs(s.imag()));
            }
    bool useRe = (mr >= mi);
    double nrm = 0.0;
    for (int i = 0; i < d1; i++)
        for (int j = 0; j < d2; j++)
            for (int k = 0; k < d3; k++) {
                double x = useRe ? Tr[i][j][k] : Ti[i][j][k];
                nrm += x * x;
            }
    nrm = std::sqrt(nrm);
    for (int i = 0; i < d1; i++)
        for (int j = 0; j < d2; j++)
            for (int k = 0; k < d3; k++)
                T[i][j][k] = (useRe ? Tr[i][j][k] : Ti[i][j][k]) / nrm;
}
} // namespace hcg

// ---------------- launch ----------------
extern "C" void kernel_launch(void* const* d_in, const int* in_sizes, int n_in,
                              void* d_out, int out_size, void* d_ws, size_t ws_size,
                              hipStream_t stream) {
    (void)in_sizes; (void)n_in; (void)out_size; (void)ws_size;
    const int*   atoms = (const int*)d_in[0];
    const float* pos   = (const float*)d_in[1];
    const int*   eidx  = (const int*)d_in[2];
    const float* emb   = (const float*)d_in[3];
    const float* f1w0  = (const float*)d_in[4];
    const float* f1b0  = (const float*)d_in[5];
    const float* f2w0  = (const float*)d_in[6];
    const float* f2b0  = (const float*)d_in[7];
    const float* f1w1  = (const float*)d_in[8];
    const float* f1b1  = (const float*)d_in[9];
    const float* f2w1  = (const float*)d_in[10];
    const float* f2b1  = (const float*)d_in[11];
    const float* pw0   = (const float*)d_in[12];
    const float* lw0   = (const float*)d_in[13];
    const float* pw1   = (const float*)d_in[14];
    const float* lw1   = (const float*)d_in[15];
    const float* encw  = (const float*)d_in[16];
    float* out = (float*)d_out;

    // workspace carve
    char* wp = (char*)d_ws;
    auto alloc = [&](size_t bytes) -> void* {
        void* r = (void*)wp;
        wp += (bytes + 255) & ~(size_t)255;
        return r;
    };
    float*    sh  = (float*)alloc((size_t)E_ * 9 * 4);
    float*    ef  = (float*)alloc((size_t)E_ * 8 * 4);
    _Float16* hid = (_Float16*)alloc((size_t)E_ * MLPC * 2);
    _Float16* Bp  = (_Float16*)alloc((size_t)2816 * MLPC * 2);   // max WN
    float*    X0  = (float*)alloc((size_t)N_ * 144 * 4);
    float*    A0  = (float*)alloc((size_t)N_ * 144 * 4);
    float*    H1  = (float*)alloc((size_t)N_ * 144 * 4);
    float*    A1  = (float*)alloc((size_t)N_ * 144 * 4);
    float*    H2  = (float*)alloc((size_t)N_ * 144 * 4);

    // host CG table (deterministic, ~microseconds)
    static const int OFFH[3] = {0, 1, 4};
    CGTab cg;
    std::memset(cg.v, 0, sizeof(cg.v));
    for (int t = 0; t < 11; t++) {
        double T[5][5][5];
        hcg::real_w3j(PL1[t], PL2[t], PL3[t], T);
        int d1 = 2 * PL1[t] + 1, d2 = 2 * PL2[t] + 1, d3 = 2 * PL3[t] + 1;
        for (int i = 0; i < d1; i++)
            for (int j = 0; j < d2; j++)
                for (int k = 0; k < d3; k++)
                    cg.v[((OFFH[PL1[t]] + i) * 9 + (OFFH[PL2[t]] + j)) * 9 +
                         (OFFH[PL3[t]] + k)] += (float)T[i][j][k];
    }

    const int WN0 = 3 * MULC * MULC, WN1 = 11 * MULC * MULC;

    // --- pipeline ---
    geom_kernel<<<(E_ + 255) / 256, 256, 0, stream>>>(pos, eidx, sh, ef);
    embed_kernel<<<(N_ * 144 + 255) / 256, 256, 0, stream>>>(atoms, emb, X0);

    // layer 0
    mlp_kernel<<<(E_ * MLPC + 255) / 256, 256, 0, stream>>>(ef, f1w0, f1b0, hid);
    packB_kernel<<<(((WN0 / 16) * 128) + 255) / 256, 256, 0, stream>>>(f2w0, Bp, WN0);
    zero_kernel<<<(N_ * 144 + 255) / 256, 256, 0, stream>>>(A0, N_ * 144);
    conv_kernel<0><<<E_ / 16, 128, 0, stream>>>(hid, Bp, f2b0, X0, sh, eidx, A0, cg);
    prod_kernel<<<N_ / 8, 128, 0, stream>>>(A0, X0, pw0, lw0, H1, cg);

    // layer 1
    mlp_kernel<<<(E_ * MLPC + 255) / 256, 256, 0, stream>>>(ef, f1w1, f1b1, hid);
    packB_kernel<<<(((WN1 / 16) * 128) + 255) / 256, 256, 0, stream>>>(f2w1, Bp, WN1);
    zero_kernel<<<(N_ * 144 + 255) / 256, 256, 0, stream>>>(A1, N_ * 144);
    conv_kernel<1><<<E_ / 16, 128, 0, stream>>>(hid, Bp, f2b1, H1, sh, eidx, A1, cg);
    prod_kernel<<<N_ / 8, 128, 0, stream>>>(A1, H1, pw1, lw1, H2, cg);

    // encoder
    enc_kernel<<<dim3(N_ / 16, ENCC / 16), 32, 0, stream>>>(H2, encw, out);
}
